// TimesBlock_moe_7971459301703
// MI455X (gfx1250) — compile-verified, matrix operand order checked by reading, not run on hardware
//
#include <hip/hip_runtime.h>
#include <hip/hip_bf16.h>
#include <math.h>

// ---------------------------------------------------------------------------
// TimesBlock MoE for MI455X (gfx1250, wave32, WMMA).
// Convs as implicit-GEMM with v_wmma_f32_16x16x32_bf16; channel-last padded
// activations -> all LDS staging is unconditional b128 traffic; double-
// buffered LDS ping-pong so global prefetch overlaps WMMA (1 barrier/step).
// ---------------------------------------------------------------------------

typedef __bf16 bf16;
typedef bf16 bf16x16 __attribute__((ext_vector_type(16)));
typedef float f32x8 __attribute__((ext_vector_type(8)));

#define B_ 32
#define T_ 96
#define N_ 231
#define F_ 49
#define COUT1 462
#define LMAX 192    // max padded sequence length (p<=96 => L<=96+p-1<192)
#define PADSP 512   // max (H+4)*(W+4) padded plane (bounded by 500)

#define CIN1P 256   // 231 padded to mult-of-32  (conv1 K channels)
#define CIN2P 480   // 462 padded to mult-of-32  (conv2 K channels)
#define CO1ALLOC 512 // conv1 out rows padded to grid 8*64
#define CO2ALLOC 256 // conv2 out rows padded to grid 4*64
#define GATE_KP 256  // gate K (231) padded
#define GATE_N  1386
#define GATE_NP 1408 // gate cols padded to 22*64

struct Meta {
  int top[2];
  int p[2];
  int H[2];
  int W[2];
  int L[2];
  int PH[2];
  int PW[2];
  float pw[B_][2];
};

union FragU { bf16x16 v; uint4 u[2]; };
union Pack8 { bf16 h[8]; uint4 u; };

// ------------------------- WMMA 32x32 per-wave core -------------------------
__device__ __forceinline__ void wmma_compute_32x32(const bf16* As, const bf16* Bs,
                                                   f32x8 acc[2][2], int lane,
                                                   int wy, int wx) {
  const int m0 = lane & 15;
  const int hi = lane >> 4;
  bf16x16 aF[2], bF[2];
#pragma unroll
  for (int r = 0; r < 2; ++r) {
    const bf16* p = As + (wy * 32 + r * 16 + m0) * 32;
    FragU f;
    f.u[0] = *(const uint4*)(p + hi * 8);       // K = hi*8..hi*8+7
    f.u[1] = *(const uint4*)(p + 16 + hi * 8);  // K = 16+hi*8..
    aF[r] = f.v;
  }
#pragma unroll
  for (int c = 0; c < 2; ++c) {
    const bf16* p = Bs + (wx * 32 + c * 16 + m0) * 32;
    FragU f;
    f.u[0] = *(const uint4*)(p + hi * 8);
    f.u[1] = *(const uint4*)(p + 16 + hi * 8);
    bF[c] = f.v;
  }
#pragma unroll
  for (int r = 0; r < 2; ++r)
#pragma unroll
    for (int c = 0; c < 2; ++c)
      acc[r][c] = __builtin_amdgcn_wmma_f32_16x16x32_bf16(
          false, aF[r], false, bF[c], (short)0, acc[r][c], false, false);
}

// ------------------------------ FFT amplitude -------------------------------
__global__ __launch_bounds__(256) void dft_amp_kernel(const float* __restrict__ x,
                                                      float* __restrict__ ampN) {
  const int b = blockIdx.x / F_;
  const int f = blockIdx.x % F_;
  __shared__ float cs[T_], sn[T_];
  __shared__ float red[256];
  const int tid = threadIdx.x;
  if (tid < T_) {
    const float ang = -2.0f * 3.14159265358979323846f * (float)f * (float)tid / (float)T_;
    cs[tid] = __cosf(ang);
    sn[tid] = __sinf(ang);
  }
  __syncthreads();
  float sum = 0.0f;
  for (int n = tid; n < N_; n += 256) {
    float re = 0.0f, im = 0.0f;
    for (int t = 0; t < T_; ++t) {
      const float v = x[((size_t)b * T_ + t) * N_ + n];
      re += v * cs[t];
      im += v * sn[t];
    }
    sum += sqrtf(re * re + im * im);
  }
  red[tid] = sum;
  __syncthreads();
  for (int s = 128; s > 0; s >>= 1) {
    if (tid < s) red[tid] += red[tid + s];
    __syncthreads();
  }
  if (tid == 0) ampN[b * F_ + f] = red[0] / (float)N_;
}

// ---------------------- top-2 periods + period weights ----------------------
__global__ void period_meta_kernel(const float* __restrict__ ampN,
                                   Meta* __restrict__ meta) {
  __shared__ float freq[F_];
  __shared__ int sTop[2];
  const int tid = threadIdx.x;
  if (tid < F_) {
    float s = 0.0f;
    for (int b = 0; b < B_; ++b) s += ampN[b * F_ + tid];
    freq[tid] = (tid == 0) ? 0.0f : s / (float)B_;
  }
  __syncthreads();
  if (tid == 0) {
    int t0 = 1; float best = freq[1];
    for (int f = 2; f < F_; ++f) if (freq[f] > best) { best = freq[f]; t0 = f; }
    int t1 = (t0 == 1) ? 2 : 1; float best1 = freq[t1];
    for (int f = 1; f < F_; ++f) {
      if (f == t0) continue;
      if (freq[f] > best1) { best1 = freq[f]; t1 = f; }
    }
    sTop[0] = t0; sTop[1] = t1;
    for (int i = 0; i < 2; ++i) {
      const int top = sTop[i];
      const int p = T_ / top;
      const int Hh = (T_ + p - 1) / p;
      meta->top[i] = top; meta->p[i] = p;
      meta->H[i] = Hh; meta->W[i] = p;
      meta->L[i] = Hh * p;
      meta->PH[i] = Hh + 4; meta->PW[i] = p + 4;
    }
  }
  __syncthreads();
  if (tid < B_) {
    const float a0 = ampN[tid * F_ + sTop[0]];
    const float a1 = ampN[tid * F_ + sTop[1]];
    const float mx = fmaxf(a0, a1);
    const float e0 = expf(a0 - mx), e1 = expf(a1 - mx);
    const float inv = 1.0f / (e0 + e1);
    meta->pw[tid][0] = e0 * inv;
    meta->pw[tid][1] = e1 * inv;
  }
}

// --------------------------- weight preparation -----------------------------
// conv1: average of {1x1,3x3,5x5} -> one effective 5x5, layout
// [CO1ALLOC][25 taps][CIN1P] bf16, zero-padded rows/channels.
__global__ void fold_conv1_kernel(const float* __restrict__ w1,
                                  const float* __restrict__ w3,
                                  const float* __restrict__ w5,
                                  bf16* __restrict__ out) {
  const int i = blockIdx.x * blockDim.x + threadIdx.x;
  if (i >= CO1ALLOC * 25 * CIN1P) return;
  const int ci = i % CIN1P;
  const int tap = (i / CIN1P) % 25;
  const int co = i / (CIN1P * 25);
  float s = 0.0f;
  if (co < COUT1 && ci < N_) {
    const int kh = tap / 5, kw = tap % 5;
    s = w5[(co * N_ + ci) * 25 + tap];
    if (kh >= 1 && kh <= 3 && kw >= 1 && kw <= 3)
      s += w3[((co * N_ + ci) * 3 + (kh - 1)) * 3 + (kw - 1)];
    if (kh == 2 && kw == 2) s += w1[co * N_ + ci];
    s *= (1.0f / 3.0f);
  }
  out[i] = (bf16)s;
}

__global__ void fold_bias_kernel(const float* __restrict__ b1,
                                 const float* __restrict__ b2,
                                 const float* __restrict__ b3,
                                 float* __restrict__ out) {
  const int i = blockIdx.x * blockDim.x + threadIdx.x;
  if (i < CO1ALLOC)
    out[i] = (i < COUT1) ? (b1[i] + b2[i] + b3[i]) * (1.0f / 3.0f) : 0.0f;
}

// conv2 weights (231,462,k,k) -> [CO2ALLOC][k2 taps][CIN2P] bf16, zero padded
__global__ void prep_conv2_kernel(const float* __restrict__ src,
                                  bf16* __restrict__ dst, int k2) {
  const int i = blockIdx.x * blockDim.x + threadIdx.x;
  if (i >= CO2ALLOC * k2 * CIN2P) return;
  const int ci = i % CIN2P;
  const int tap = (i / CIN2P) % k2;
  const int co = i / (CIN2P * k2);
  float v = 0.0f;
  if (co < N_ && ci < COUT1) v = src[(co * COUT1 + ci) * k2 + tap];
  dst[i] = (bf16)v;
}

// gate (n,e,s) -> transposed padded [GATE_NP][GATE_KP] bf16
__global__ void prep_gateT_kernel(const float* __restrict__ gate,
                                  bf16* __restrict__ dst) {
  const int i = blockIdx.x * blockDim.x + threadIdx.x;
  if (i >= GATE_NP * GATE_KP) return;
  const int k = i % GATE_KP;
  const int col = i / GATE_KP;
  float v = 0.0f;
  if (col < GATE_N && k < N_) v = gate[(size_t)k * GATE_N + col];
  dst[i] = (bf16)v;
}

// x (B,T,N) -> padded A [3072][GATE_KP] bf16
__global__ void prep_xpad_kernel(const float* __restrict__ x,
                                 bf16* __restrict__ dst) {
  const int i = blockIdx.x * blockDim.x + threadIdx.x;
  if (i >= B_ * T_ * GATE_KP) return;
  const int k = i % GATE_KP;
  const int row = i / GATE_KP;
  dst[i] = (bf16)((k < N_) ? x[(size_t)row * N_ + k] : 0.0f);
}

__global__ void zero_u32_kernel(unsigned* __restrict__ p, int n) {
  const int i = blockIdx.x * blockDim.x + threadIdx.x;
  if (i < n) p[i] = 0u;
}

// ------- pack (B,T,N) -> zero-haloed channel-last (B,PH,PW,CIN1P) bf16 ------
__global__ void pack_input_kernel(const float* __restrict__ src,
                                  bf16* __restrict__ dstPad,
                                  const Meta* __restrict__ meta, int period) {
  const int W = meta->W[period];
  const int L = meta->L[period];
  const int PH = meta->PH[period];
  const int PW = meta->PW[period];
  const int idx = blockIdx.x * blockDim.x + threadIdx.x;
  if (idx >= B_ * L * N_) return;
  const int n = idx % N_;
  const int rest = idx / N_;
  const int pos = rest % L;
  const int b = rest / L;
  const float v = (pos < T_) ? src[((size_t)b * T_ + pos) * N_ + n] : 0.0f;
  const int h = pos / W;
  const int w = pos - h * W;
  dstPad[(((size_t)b * PH + h + 2) * PW + (w + 2)) * CIN1P + n] = (bf16)v;
}

// ------------------- implicit-GEMM conv via WMMA (bf16->f32) ----------------
// C[co][m] = sum over taps,ci of Wt[co][tap][ci] * inPad[spatial(m,tap)][ci]
// Double-buffered LDS pipeline: prefetch tile(it+1) to regs, wmma on buf[it&1].
// mode 0: +bias, exact GELU -> bf16 channel-last padded out (vector store)
// mode 1: +bias -> experts[eIdx][b][co][t] fp32 (t<96)
__global__ __launch_bounds__(128) void conv_gemm_kernel(
    const bf16* __restrict__ inPad,  // (B,PH,PW,cinPad) channel-last
    const bf16* __restrict__ Wt,     // (coAlloc, ks2, cinPad)
    const float* __restrict__ bias,
    const Meta* __restrict__ meta, int period,
    int cinPad, int ks, int mode, int coutStore,
    bf16* __restrict__ outPad, int outCPad,
    float* __restrict__ experts, int eIdx) {
  const int W = meta->W[period];
  const int L = meta->L[period];
  const int PH = meta->PH[period];
  const int PW = meta->PW[period];
  const int Mtot = B_ * L;
  const int mBase = blockIdx.x * 64;
  if (mBase >= Mtot) return;  // uniform exit: EXEC all-ones at WMMA
  const int coBase = blockIdx.y * 64;
  const int ks2 = ks * ks;
  const size_t Ktot = (size_t)ks2 * cinPad;
  const int roff = 2 - (ks >> 1);

  __shared__ __align__(16) uint4 AsB[2][64 * 4];  // ping-pong 64x32 bf16 tiles
  __shared__ __align__(16) uint4 BsB[2][64 * 4];

  const int tid = threadIdx.x;
  const int lane = tid & 31;
  const int wave = tid >> 5;
  const int wy = wave >> 1;
  const int wx = wave & 1;
  const int srow = tid >> 1;   // 0..63
  const int shalf = tid & 1;   // 16-element half
  const int dstIdx = srow * 4 + shalf * 2;

  // A stage source (rows zero-padded at prep; [co][tap][ci] => purely linear)
  const bf16* aP = Wt + (size_t)(coBase + srow) * Ktot + shalf * 16;
  // B stage source: clamp OOB columns to a valid address (their C columns are
  // never stored, so garbage is harmless and staging stays branch-free)
  const int gm = mBase + srow;
  const int gmc = (gm < Mtot) ? gm : (Mtot - 1);
  const int gb = gmc / L;
  const int gpos = gmc - gb * L;
  const int gh = gpos / W;
  const int gw = gpos - gh * W;
  const bf16* bTap =
      inPad + (((size_t)gb * PH + gh + roff) * PW + (gw + roff)) * cinPad + shalf * 16;
  const bf16* bP = bTap;
  int kc = 0, kwi = 0;

  // incremental tap/channel walk (uniform control, no divisions in the loop)
  auto advance = [&]() {
    aP += 32;
    kc += 32;
    if (kc == cinPad) {
      kc = 0;
      ++kwi;
      bTap += cinPad;                                // next kw
      if (kwi == ks) { kwi = 0; bTap += ((size_t)PW - ks) * cinPad; }  // next kh
      bP = bTap;
    } else {
      bP += 32;
    }
  };

  const f32x8 zero = {0.f, 0.f, 0.f, 0.f, 0.f, 0.f, 0.f, 0.f};
  f32x8 acc[2][2];
#pragma unroll
  for (int r = 0; r < 2; ++r)
#pragma unroll
    for (int c = 0; c < 2; ++c) acc[r][c] = zero;

  const int nIter = ks2 * (cinPad >> 5);

  // preload tile 0 into buffer 0
  {
    const uint4 a0 = *(const uint4*)(aP);
    const uint4 a1 = *(const uint4*)(aP + 8);
    const uint4 b0 = *(const uint4*)(bP);
    const uint4 b1 = *(const uint4*)(bP + 8);
    AsB[0][dstIdx] = a0; AsB[0][dstIdx + 1] = a1;
    BsB[0][dstIdx] = b0; BsB[0][dstIdx + 1] = b1;
    advance();
  }

  for (int it = 0; it < nIter; ++it) {
    uint4 a0, a1, b0, b1;
    const bool more = (it + 1) < nIter;
    if (more) {  // uniform predicate; global prefetch overlaps wmma below
      a0 = *(const uint4*)(aP);
      a1 = *(const uint4*)(aP + 8);
      b0 = *(const uint4*)(bP);
      b1 = *(const uint4*)(bP + 8);
      advance();
    }
    __syncthreads();  // buf[it&1] tile visible; prior readers of buf[next] done
    wmma_compute_32x32((const bf16*)AsB[it & 1], (const bf16*)BsB[it & 1],
                       acc, lane, wy, wx);
    if (more) {
      const int nb = (it + 1) & 1;
      AsB[nb][dstIdx] = a0; AsB[nb][dstIdx + 1] = a1;
      BsB[nb][dstIdx] = b0; BsB[nb][dstIdx + 1] = b1;
    }
  }

  // epilogue: lane holds col = lane&15, rows i + (lane>>4)*8 per acc element
  const int m0 = lane & 15;
  const int hi = lane >> 4;
#pragma unroll
  for (int c = 0; c < 2; ++c) {
    const int m = mBase + wx * 32 + c * 16 + m0;
    if (m >= Mtot) continue;
    const int b = m / L;
    const int pos = m - b * L;
    if (mode == 0) {
      const int h = pos / W;
      const int w = pos - h * W;
      const size_t spatial = ((size_t)b * PH + h + 2) * PW + (w + 2);
#pragma unroll
      for (int r = 0; r < 2; ++r) {
        const int co0 = coBase + wy * 32 + r * 16 + hi * 8;
        if (co0 >= coutStore) continue;
        Pack8 pk;
#pragma unroll
        for (int i = 0; i < 8; ++i) {
          const float v = acc[r][c][i] + bias[co0 + i];
          pk.h[i] = (bf16)(0.5f * v * (1.0f + erff(v * 0.70710678118654752f)));
        }
        *(uint4*)(outPad + spatial * outCPad + co0) = pk.u;  // 16B aligned
      }
    } else {
      if (pos >= T_) continue;
#pragma unroll
      for (int r = 0; r < 2; ++r) {
#pragma unroll
        for (int i = 0; i < 8; ++i) {
          const int co = coBase + wy * 32 + r * 16 + hi * 8 + i;
          if (co < coutStore)
            experts[(((size_t)eIdx * B_ + b) * N_ + co) * T_ + pos] =
                acc[r][c][i] + bias[co];
        }
      }
    }
  }
}

// ----------------------- dense GEMM (gate logits) via WMMA ------------------
// C[M][GATE_N] = A[M][GATE_KP] * BT[GATE_NP][GATE_KP]^T, double-buffered.
__global__ __launch_bounds__(128) void dense_gemm_kernel(
    const bf16* __restrict__ A, const bf16* __restrict__ BT,
    float* __restrict__ C) {
  const int rowBase = blockIdx.y * 64;
  const int colBase = blockIdx.x * 64;
  __shared__ __align__(16) uint4 AsB[2][64 * 4];
  __shared__ __align__(16) uint4 BsB[2][64 * 4];
  const int tid = threadIdx.x;
  const int lane = tid & 31;
  const int wave = tid >> 5;
  const int wy = wave >> 1;
  const int wx = wave & 1;
  const int srow = tid >> 1;
  const int shalf = tid & 1;
  const int dstIdx = srow * 4 + shalf * 2;

  const bf16* aP = A + (size_t)(rowBase + srow) * GATE_KP + shalf * 16;
  const bf16* bP = BT + (size_t)(colBase + srow) * GATE_KP + shalf * 16;

  const f32x8 zero = {0.f, 0.f, 0.f, 0.f, 0.f, 0.f, 0.f, 0.f};
  f32x8 acc[2][2];
#pragma unroll
  for (int r = 0; r < 2; ++r)
#pragma unroll
    for (int c = 0; c < 2; ++c) acc[r][c] = zero;

  const int nIter = GATE_KP / 32;
  {
    const uint4 a0 = *(const uint4*)(aP);
    const uint4 a1 = *(const uint4*)(aP + 8);
    const uint4 b0 = *(const uint4*)(bP);
    const uint4 b1 = *(const uint4*)(bP + 8);
    AsB[0][dstIdx] = a0; AsB[0][dstIdx + 1] = a1;
    BsB[0][dstIdx] = b0; BsB[0][dstIdx + 1] = b1;
    aP += 32; bP += 32;
  }
  for (int it = 0; it < nIter; ++it) {
    uint4 a0, a1, b0, b1;
    const bool more = (it + 1) < nIter;
    if (more) {
      a0 = *(const uint4*)(aP);
      a1 = *(const uint4*)(aP + 8);
      b0 = *(const uint4*)(bP);
      b1 = *(const uint4*)(bP + 8);
      aP += 32; bP += 32;
    }
    __syncthreads();
    wmma_compute_32x32((const bf16*)AsB[it & 1], (const bf16*)BsB[it & 1],
                       acc, lane, wy, wx);
    if (more) {
      const int nb = (it + 1) & 1;
      AsB[nb][dstIdx] = a0; AsB[nb][dstIdx + 1] = a1;
      BsB[nb][dstIdx] = b0; BsB[nb][dstIdx + 1] = b1;
    }
  }

  const int m0 = lane & 15;
  const int hi = lane >> 4;
#pragma unroll
  for (int c = 0; c < 2; ++c) {
    const int col = colBase + wx * 32 + c * 16 + m0;
    if (col >= GATE_N) continue;
#pragma unroll
    for (int r = 0; r < 2; ++r) {
#pragma unroll
      for (int i = 0; i < 8; ++i) {
        const int row = rowBase + wy * 32 + r * 16 + hi * 8 + i;
        C[(size_t)row * GATE_N + col] = acc[r][c][i];  // rows always < 3072
      }
    }
  }
}

// --------------------- per-location softmax over experts --------------------
__global__ void gate_softmax_kernel(float* __restrict__ logits) {
  const int i = blockIdx.x * blockDim.x + threadIdx.x;
  if (i >= B_ * T_ * N_) return;
  const int s = i % N_;
  const int m = i / N_;
  float* row = logits + (size_t)m * GATE_N + s;
  float v[6];
  float mx = -1e30f;
#pragma unroll
  for (int e = 0; e < 6; ++e) { v[e] = row[e * N_]; mx = fmaxf(mx, v[e]); }
  float sum = 0.0f;
#pragma unroll
  for (int e = 0; e < 6; ++e) { v[e] = expf(v[e] - mx); sum += v[e]; }
  const float inv = 1.0f / sum;
#pragma unroll
  for (int e = 0; e < 6; ++e) row[e * N_] = v[e] * inv;
}

// ------------------------- MoE combine + period mix -------------------------
__global__ void combine_kernel(const float* __restrict__ wsm,
                               const float* __restrict__ experts,
                               const Meta* __restrict__ meta, int period,
                               float* __restrict__ out) {
  const int i = blockIdx.x * blockDim.x + threadIdx.x;
  if (i >= B_ * T_ * N_) return;
  const int n = i % N_;
  const int m = i / N_;
  const int t = m % T_;
  const int b = m / T_;
  const float pwv = meta->pw[b][period];
  float acc = 0.0f;
#pragma unroll
  for (int e = 0; e < 6; ++e)
    acc += wsm[(size_t)m * GATE_N + e * N_ + n] *
           experts[(((size_t)e * B_ + b) * N_ + n) * T_ + t];
  if (period == 0) out[i] = pwv * acc;
  else out[i] += pwv * acc;
}

// ---------------------------------------------------------------------------
extern "C" void kernel_launch(void* const* d_in, const int* in_sizes, int n_in,
                              void* d_out, int out_size, void* d_ws, size_t ws_size,
                              hipStream_t stream) {
  (void)in_sizes; (void)n_in; (void)out_size; (void)ws_size;
  // 0:x 1:y 2-4:convx1_w 5-7:convx1_b 8-10:convx2_w 11-13:convx2_b
  // 14-16:convy1_w 17-19:convy1_b 20-22:convy2_w 23-25:convy2_b 26:gate_w
  const float* x = (const float*)d_in[0];
  const float* y = (const float*)d_in[1];
  const float* cx1w[3] = {(const float*)d_in[2], (const float*)d_in[3], (const float*)d_in[4]};
  const float* cx1b[3] = {(const float*)d_in[5], (const float*)d_in[6], (const float*)d_in[7]};
  const float* cx2w[3] = {(const float*)d_in[8], (const float*)d_in[9], (const float*)d_in[10]};
  const float* cx2b[3] = {(const float*)d_in[11], (const float*)d_in[12], (const float*)d_in[13]};
  const float* cy1w[3] = {(const float*)d_in[14], (const float*)d_in[15], (const float*)d_in[16]};
  const float* cy1b[3] = {(const float*)d_in[17], (const float*)d_in[18], (const float*)d_in[19]};
  const float* cy2w[3] = {(const float*)d_in[20], (const float*)d_in[21], (const float*)d_in[22]};
  const float* cy2b[3] = {(const float*)d_in[23], (const float*)d_in[24], (const float*)d_in[25]};
  const float* gate = (const float*)d_in[26];
  float* out = (float*)d_out;

  // ---- workspace carve-out ----
  char* ws = (char*)d_ws;
  size_t off = 0;
  auto take = [&](size_t bytes) { size_t r = off; off += (bytes + 255) & ~(size_t)255; return r; };
  float* ampN = (float*)(ws + take((size_t)B_ * F_ * 4));
  Meta* meta = (Meta*)(ws + take(sizeof(Meta)));
  bf16* w1x = (bf16*)(ws + take((size_t)CO1ALLOC * 25 * CIN1P * 2));
  bf16* w1y = (bf16*)(ws + take((size_t)CO1ALLOC * 25 * CIN1P * 2));
  float* b1x = (float*)(ws + take((size_t)CO1ALLOC * 4));
  float* b1y = (float*)(ws + take((size_t)CO1ALLOC * 4));
  bf16* w2x[3]; bf16* w2y[3];
  const int k2n[3] = {1, 9, 25};
  for (int k = 0; k < 3; ++k) w2x[k] = (bf16*)(ws + take((size_t)CO2ALLOC * k2n[k] * CIN2P * 2));
  for (int k = 0; k < 3; ++k) w2y[k] = (bf16*)(ws + take((size_t)CO2ALLOC * k2n[k] * CIN2P * 2));
  bf16* gateBT = (bf16*)(ws + take((size_t)GATE_NP * GATE_KP * 2));
  bf16* xbf = (bf16*)(ws + take((size_t)B_ * T_ * GATE_KP * 2));
  float* logits = (float*)(ws + take((size_t)B_ * T_ * GATE_N * 4));
  bf16* pad1 = (bf16*)(ws + take((size_t)B_ * PADSP * CIN1P * 2));   // conv1 in
  bf16* c1pad = (bf16*)(ws + take((size_t)B_ * PADSP * CIN2P * 2));  // conv1 out
  float* experts = (float*)(ws + take((size_t)6 * B_ * N_ * T_ * 4));

  const int BT_N = B_ * T_ * N_;
  const int pad1U32 = (int)((size_t)B_ * PADSP * CIN1P * 2 / 4);
  const int c1U32 = (int)((size_t)B_ * PADSP * CIN2P * 2 / 4);
  auto cdiv = [](int a, int b) { return (a + b - 1) / b; };

  // ---- stage 1: FFT amplitudes + period meta ----
  dft_amp_kernel<<<dim3(B_ * F_), dim3(256), 0, stream>>>(x, ampN);
  period_meta_kernel<<<dim3(1), dim3(64), 0, stream>>>(ampN, meta);

  // ---- stage 2: weight prep (padded, pre-swizzled, bf16) ----
  {
    const int nf = CO1ALLOC * 25 * CIN1P;
    fold_conv1_kernel<<<cdiv(nf, 256), 256, 0, stream>>>(cx1w[0], cx1w[1], cx1w[2], w1x);
    fold_conv1_kernel<<<cdiv(nf, 256), 256, 0, stream>>>(cy1w[0], cy1w[1], cy1w[2], w1y);
    fold_bias_kernel<<<cdiv(CO1ALLOC, 256), 256, 0, stream>>>(cx1b[0], cx1b[1], cx1b[2], b1x);
    fold_bias_kernel<<<cdiv(CO1ALLOC, 256), 256, 0, stream>>>(cy1b[0], cy1b[1], cy1b[2], b1y);
    for (int k = 0; k < 3; ++k) {
      const int n = CO2ALLOC * k2n[k] * CIN2P;
      prep_conv2_kernel<<<cdiv(n, 256), 256, 0, stream>>>(cx2w[k], w2x[k], k2n[k]);
      prep_conv2_kernel<<<cdiv(n, 256), 256, 0, stream>>>(cy2w[k], w2y[k], k2n[k]);
    }
    prep_gateT_kernel<<<cdiv(GATE_NP * GATE_KP, 256), 256, 0, stream>>>(gate, gateBT);
    prep_xpad_kernel<<<cdiv(B_ * T_ * GATE_KP, 256), 256, 0, stream>>>(x, xbf);
  }

  // ---- stage 3: gate logits GEMM + softmax over experts ----
  dense_gemm_kernel<<<dim3(GATE_NP / 64, (B_ * T_) / 64), 128, 0, stream>>>(
      xbf, gateBT, logits);
  gate_softmax_kernel<<<cdiv(BT_N, 256), 256, 0, stream>>>(logits);

  // ---- stage 4: per-period conv pipelines (implicit-GEMM WMMA) ----
  const dim3 convGrid1(cdiv(B_ * LMAX, 64), CO1ALLOC / 64);  // 96 x 8
  const dim3 convGrid2(cdiv(B_ * LMAX, 64), CO2ALLOC / 64);  // 96 x 4
  const int packGrid = cdiv(B_ * LMAX * N_, 256);
  const int ksArr[3] = {1, 3, 5};

  for (int period = 0; period < 2; ++period) {
    // ---- x path -> experts 0..2 ----
    zero_u32_kernel<<<cdiv(pad1U32, 256), 256, 0, stream>>>((unsigned*)pad1, pad1U32);
    pack_input_kernel<<<packGrid, 256, 0, stream>>>(x, pad1, meta, period);
    zero_u32_kernel<<<cdiv(c1U32, 256), 256, 0, stream>>>((unsigned*)c1pad, c1U32);
    conv_gemm_kernel<<<convGrid1, 128, 0, stream>>>(
        pad1, w1x, b1x, meta, period, CIN1P, 5, 0, CIN2P, c1pad, CIN2P, nullptr, 0);
    for (int k = 0; k < 3; ++k)
      conv_gemm_kernel<<<convGrid2, 128, 0, stream>>>(
          c1pad, w2x[k], cx2b[k], meta, period, CIN2P, ksArr[k], 1, N_,
          nullptr, 0, experts, k);
    // ---- y path -> experts 3..5 ----
    zero_u32_kernel<<<cdiv(pad1U32, 256), 256, 0, stream>>>((unsigned*)pad1, pad1U32);
    pack_input_kernel<<<packGrid, 256, 0, stream>>>(y, pad1, meta, period);
    zero_u32_kernel<<<cdiv(c1U32, 256), 256, 0, stream>>>((unsigned*)c1pad, c1U32);
    conv_gemm_kernel<<<convGrid1, 128, 0, stream>>>(
        pad1, w1y, b1y, meta, period, CIN1P, 5, 0, CIN2P, c1pad, CIN2P, nullptr, 0);
    for (int k = 0; k < 3; ++k)
      conv_gemm_kernel<<<convGrid2, 128, 0, stream>>>(
          c1pad, w2y[k], cy2b[k], meta, period, CIN2P, ksArr[k], 1, N_,
          nullptr, 0, experts, 3 + k);
    // ---- MoE combine weighted by softmaxed period weight ----
    combine_kernel<<<cdiv(BT_N, 256), 256, 0, stream>>>(logits, experts, meta, period, out);
  }
}
